// RefPagedPrefillGQABlock_60550448939421
// MI455X (gfx1250) — matrix operand order repaired and checked by reading, not compiled, not run on hardware
//
#include <hip/hip_runtime.h>
#include <hip/hip_bf16.h>

// ---------------- problem constants ----------------
#define B_      2
#define S_      2048
#define H_      32
#define HKV_    8
#define D_      128
#define BS_     16
#define HIDDEN_ 4096
#define G_      4
#define NB_     128
#define T_      (B_ * S_)        // 4096
#define NQKV_   (HIDDEN_ + 2 * HKV_ * D_)   // 6144
#define SCALE_  0.08838834764831845f        // 128^-0.5
#define NEG_INF_ (-1.0e30f)

typedef __attribute__((ext_vector_type(16))) __bf16 v16bf;
typedef __attribute__((ext_vector_type(8)))  __bf16 v8bf;
typedef __attribute__((ext_vector_type(8)))  float  v8f;
typedef int v4i32 __attribute__((vector_size(16)));   // matches async-LDS builtin param

static __device__ __forceinline__ v16bf cat16(v8bf a, v8bf b) {
    return __builtin_shufflevector(a, b, 0,1,2,3,4,5,6,7,8,9,10,11,12,13,14,15);
}

static __device__ __forceinline__ v8f wmma_bf16(v16bf a, v16bf b, v8f c) {
    // (neg_a, A, neg_b, B, c_mod, C, reuse_a, reuse_b)
    return __builtin_amdgcn_wmma_f32_16x16x32_bf16(false, a, false, b, (short)0, c, false, false);
}

// ---- gfx1250 async global->LDS path (probe-guarded; falls back to sync) ----
#if defined(__has_builtin)
#if __has_builtin(__builtin_amdgcn_global_load_async_to_lds_b128) && \
    __has_builtin(__builtin_amdgcn_s_wait_asynccnt)
#define ASYNC_LDS 1
#endif
#endif

#ifdef ASYNC_LDS
static __device__ __forceinline__ void async_copy16(void* lds, const void* g) {
    __builtin_amdgcn_global_load_async_to_lds_b128(
        (__attribute__((address_space(1))) v4i32*)(g),
        (__attribute__((address_space(3))) v4i32*)(lds), 0, 0);
}
static __device__ __forceinline__ void async_wait0() {
    __builtin_amdgcn_s_wait_asynccnt(0);
}
#else
static __device__ __forceinline__ void async_copy16(void* lds, const void* g) {
    *(v8bf*)lds = *(const v8bf*)g;
}
static __device__ __forceinline__ void async_wait0() {}
#endif

// =====================================================================
// Kernel 1: QKV projection.  hidden[T,4096] (f32) x w_qkv[4096,6144] (f32)
// -> q_ws[T,4096] bf16 ; k/v scattered through block_tables into paged
//    bf16 caches  k_ws/v_ws [B*NB, HKV, BS, D].
// Block: 256 thr (8 waves). Tile: 128(M) x 128(N), K-step 32.
// Per wave per K-step: 1 A-frag, 8 B-frags, 8 WMMAs.
// =====================================================================
__global__ __launch_bounds__(256)
void qkv_gemm_kernel(const float* __restrict__ hidden,
                     const float* __restrict__ w,
                     const int*   __restrict__ bt,
                     __bf16* __restrict__ q_ws,
                     __bf16* __restrict__ k_ws,
                     __bf16* __restrict__ v_ws)
{
    __shared__ __bf16 As[128 * 32];   // [row][k]            8 KB
    __shared__ __bf16 Bs[128 * 32];   // [n][k] (transposed) 8 KB

    const int tid  = threadIdx.x;
    const int lane = tid & 31;
    const int wave = tid >> 5;        // 0..7 : 16-row strip within the 128-row tile
    const int half = lane >> 4;       // 0/1
    const int lm   = lane & 15;

    const int row0 = blockIdx.y * 128;   // token rows
    const int col0 = blockIdx.x * 128;   // qkv cols

    v8f acc[8] = {};                     // 8 col-tiles of 16

    for (int k0 = 0; k0 < HIDDEN_; k0 += 32) {
        __syncthreads();
        // ---- stage A tile 128x32 f32 -> bf16 (float4 per thread, x4) ----
        #pragma unroll
        for (int it = 0; it < 4; ++it) {
            int e = (it * 256 + tid) * 4;
            int r = e >> 5, c = e & 31;
            float4 f = *(const float4*)(hidden + (size_t)(row0 + r) * HIDDEN_ + k0 + c);
            __bf16* dst = &As[r * 32 + c];
            dst[0] = (__bf16)f.x; dst[1] = (__bf16)f.y;
            dst[2] = (__bf16)f.z; dst[3] = (__bf16)f.w;
        }
        // ---- stage B tile 32x128 f32 -> bf16 transposed into Bs[n][k] ----
        #pragma unroll
        for (int it = 0; it < 4; ++it) {
            int e = (it * 256 + tid) * 4;
            int kr = e >> 7, c = e & 127;
            float4 f = *(const float4*)(w + (size_t)(k0 + kr) * NQKV_ + col0 + c);
            Bs[(c + 0) * 32 + kr] = (__bf16)f.x;
            Bs[(c + 1) * 32 + kr] = (__bf16)f.y;
            Bs[(c + 2) * 32 + kr] = (__bf16)f.z;
            Bs[(c + 3) * 32 + kr] = (__bf16)f.w;
        }
        __syncthreads();

        if (k0 + 32 < HIDDEN_) {   // pull next tiles toward the caches
            __builtin_prefetch(hidden + (size_t)(row0 + (tid & 127)) * HIDDEN_ + k0 + 32, 0, 1);
            __builtin_prefetch(w + (size_t)(k0 + 32 + (tid & 31)) * NQKV_ + col0, 0, 1);
        }

        // ---- A fragment (16x32 bf16): lane M = lm, half selects K-subrange ----
        const __bf16* ap = &As[(wave * 16 + lm) * 32 + half * 8];
        v16bf afrag = cat16(*(const v8bf*)ap, *(const v8bf*)(ap + 16));

        #pragma unroll
        for (int t = 0; t < 8; ++t) {
            const __bf16* bp = &Bs[(t * 16 + lm) * 32 + half * 8];
            v16bf bfrag = cat16(*(const v8bf*)bp, *(const v8bf*)(bp + 16));
            acc[t] = wmma_bf16(afrag, bfrag, acc[t]);
        }
    }

    // ---- epilogue: route q / paged-k / paged-v ----
    #pragma unroll
    for (int t = 0; t < 8; ++t) {
        #pragma unroll
        for (int r = 0; r < 8; ++r) {
            int grow = row0 + wave * 16 + half * 8 + r;   // token
            int gcol = col0 + t * 16 + lm;                // qkv column
            __bf16 bv = (__bf16)acc[t][r];
            if (gcol < HIDDEN_) {
                q_ws[(size_t)grow * HIDDEN_ + gcol] = bv;
            } else {
                int kv  = gcol - HIDDEN_;
                int isv = (kv >= HKV_ * D_);
                int kvc = isv ? kv - HKV_ * D_ : kv;
                int h = kvc / D_, d = kvc % D_;
                int b = grow / S_, s = grow % S_;
                int page = bt[b * NB_ + (s >> 4)];
                size_t idx = (((size_t)page * HKV_ + h) * BS_ + (s & 15)) * D_ + d;
                (isv ? v_ws : k_ws)[idx] = bv;
            }
        }
    }
}

// =====================================================================
// Kernel 2: causal GQA flash attention over the paged bf16 caches.
// Block = (b, kv-head, 16-query tile); 4 waves = the 4 query heads of the
// group (identical causal range -> shared K/V LDS staging).
// =====================================================================
__global__ __launch_bounds__(128)
void attn_kernel(const __bf16* __restrict__ q_ws,
                 const __bf16* __restrict__ k_ws,
                 const __bf16* __restrict__ v_ws,
                 const int*    __restrict__ bt,
                 __bf16* __restrict__ attn_ws)
{
    __shared__ __bf16 Ks[32 * 128];       // [key][d]        8 KB
    __shared__ __bf16 Vs[128 * 32];       // [d][key] (T)    8 KB
    __shared__ __bf16 Ps[4][16 * 32];     // per-wave P      4 KB

    const int tid  = threadIdx.x;
    const int lane = tid & 31;
    const int wave = tid >> 5;            // g (query head in group)
    const int half = lane >> 4;
    const int lm   = lane & 15;

    const int qt = blockIdx.x;            // 16-query tile 0..127
    const int bh = blockIdx.y;            // b*HKV + h
    const int b  = bh / HKV_, h = bh % HKV_;
    const int hq = h * G_ + wave;
    const int qrow0 = qt * 16;

    // ---- preload Q fragments (16 rows x D=128, four K=32 chunks) ----
    v16bf qfrag[4];
    const __bf16* qbase = q_ws + (size_t)(b * S_ + qrow0 + lm) * HIDDEN_ + hq * D_;
    #pragma unroll
    for (int c = 0; c < 4; ++c) {
        const __bf16* qp = qbase + c * 32 + half * 8;
        qfrag[c] = cat16(*(const v8bf*)qp, *(const v8bf*)(qp + 16));
    }

    v8f   oacc[8] = {};
    float mrow[8], lrow[8];
    #pragma unroll
    for (int r = 0; r < 8; ++r) { mrow[r] = NEG_INF_; lrow[r] = 0.f; }

    const int nkb = (qrow0 + 15) / 32 + 1;    // 32-key blocks needed (causal)

    for (int kb = 0; kb < nkb; ++kb) {
        __syncthreads();
        // ---- cooperative stage: K via async global->LDS, V transposed ----
        #pragma unroll
        for (int it = 0; it < 4; ++it) {
            int e  = (it * 128 + tid) * 8;
            int kk = e >> 7, d = e & 127;
            int kg = kb * 32 + kk;
            int page = bt[b * NB_ + (kg >> 4)];
            size_t base = (((size_t)page * HKV_ + h) * BS_ + (kg & 15)) * D_ + d;
            async_copy16(&Ks[kk * 128 + d], k_ws + base);     // ASYNCcnt path
            v8bf vv = *(const v8bf*)(v_ws + base);
            #pragma unroll
            for (int j = 0; j < 8; ++j) Vs[(d + j) * 32 + kk] = vv[j];
        }
        async_wait0();
        __syncthreads();

        // ---- scores: two 16(key)-tiles, contraction over D=128 ----
        v8f sacc[2] = {};
        #pragma unroll
        for (int u = 0; u < 2; ++u) {
            #pragma unroll
            for (int c = 0; c < 4; ++c) {
                const __bf16* kp = &Ks[(u * 16 + lm) * 128 + c * 32 + half * 8];
                v16bf kfrag = cat16(*(const v8bf*)kp, *(const v8bf*)(kp + 16));
                sacc[u] = wmma_bf16(qfrag[c], kfrag, sacc[u]);
            }
        }

        // ---- causal mask + online softmax (rows live across 16-lane halves) ----
        #pragma unroll
        for (int r = 0; r < 8; ++r) {
            int grow = qrow0 + half * 8 + r;
            int key0 = kb * 32 + lm;
            float s0 = sacc[0][r] * SCALE_;
            float s1 = sacc[1][r] * SCALE_;
            if (key0 > grow)      s0 = NEG_INF_;
            if (key0 + 16 > grow) s1 = NEG_INF_;
            float mx = fmaxf(s0, s1);
            #pragma unroll
            for (int off = 1; off < 16; off <<= 1)
                mx = fmaxf(mx, __shfl_xor(mx, off, 32));
            float mnew  = fmaxf(mrow[r], mx);
            float alpha = __expf(mrow[r] - mnew);
            float p0 = __expf(s0 - mnew);
            float p1 = __expf(s1 - mnew);
            float ls = p0 + p1;
            #pragma unroll
            for (int off = 1; off < 16; off <<= 1)
                ls += __shfl_xor(ls, off, 32);
            lrow[r] = lrow[r] * alpha + ls;
            mrow[r] = mnew;
            #pragma unroll
            for (int t = 0; t < 8; ++t) oacc[t][r] *= alpha;
            // park P (C-layout) into LDS for A-layout reload
            Ps[wave][(half * 8 + r) * 32 + lm]      = (__bf16)p0;
            Ps[wave][(half * 8 + r) * 32 + 16 + lm] = (__bf16)p1;
        }

        // ---- reload P as an A fragment (16x32), then PV ----
        const __bf16* pp = &Ps[wave][lm * 32 + half * 8];
        v16bf pfrag = cat16(*(const v8bf*)pp, *(const v8bf*)(pp + 16));
        #pragma unroll
        for (int t = 0; t < 8; ++t) {
            const __bf16* vp = &Vs[(t * 16 + lm) * 32 + half * 8];
            v16bf vfrag = cat16(*(const v8bf*)vp, *(const v8bf*)(vp + 16));
            oacc[t] = wmma_bf16(pfrag, vfrag, oacc[t]);
        }
    }

    // ---- normalize and store token-major [B,S,H,D] as bf16 ----
    #pragma unroll
    for (int t = 0; t < 8; ++t) {
        #pragma unroll
        for (int r = 0; r < 8; ++r) {
            int row = qrow0 + half * 8 + r;
            float o = oacc[t][r] / lrow[r];
            attn_ws[(((size_t)(b * S_ + row)) * H_ + hq) * D_ + t * 16 + lm] = (__bf16)o;
        }
    }
}

// =====================================================================
// Kernel 3: output projection.  attn_ws[T,4096] bf16 x w_out[4096,4096] f32
// -> d_out[T,4096] f32.  Tile 128x128, A staged via async global->LDS.
// =====================================================================
__global__ __launch_bounds__(256)
void out_gemm_kernel(const __bf16* __restrict__ A,
                     const float*  __restrict__ w,
                     float* __restrict__ out)
{
    __shared__ __bf16 As[128 * 32];
    __shared__ __bf16 Bs[128 * 32];

    const int tid  = threadIdx.x;
    const int lane = tid & 31;
    const int wave = tid >> 5;
    const int half = lane >> 4;
    const int lm   = lane & 15;

    const int row0 = blockIdx.y * 128;
    const int col0 = blockIdx.x * 128;

    v8f acc[8] = {};

    for (int k0 = 0; k0 < HIDDEN_; k0 += 32) {
        __syncthreads();
        // A already bf16: direct async global->LDS copies (16B per lane, x2)
        #pragma unroll
        for (int it = 0; it < 2; ++it) {
            int e = (it * 256 + tid) * 8;
            int r = e >> 5, c = e & 31;
            async_copy16(&As[r * 32 + c],
                         A + (size_t)(row0 + r) * HIDDEN_ + k0 + c);
        }
        // B tile 32x128 f32 -> bf16 transposed
        #pragma unroll
        for (int it = 0; it < 4; ++it) {
            int e = (it * 256 + tid) * 4;
            int kr = e >> 7, c = e & 127;
            float4 f = *(const float4*)(w + (size_t)(k0 + kr) * HIDDEN_ + col0 + c);
            Bs[(c + 0) * 32 + kr] = (__bf16)f.x;
            Bs[(c + 1) * 32 + kr] = (__bf16)f.y;
            Bs[(c + 2) * 32 + kr] = (__bf16)f.z;
            Bs[(c + 3) * 32 + kr] = (__bf16)f.w;
        }
        async_wait0();
        __syncthreads();

        if (k0 + 32 < HIDDEN_) {
            __builtin_prefetch(A + (size_t)(row0 + (tid & 127)) * HIDDEN_ + k0 + 32, 0, 1);
            __builtin_prefetch(w + (size_t)(k0 + 32 + (tid & 31)) * HIDDEN_ + col0, 0, 1);
        }

        const __bf16* ap = &As[(wave * 16 + lm) * 32 + half * 8];
        v16bf afrag = cat16(*(const v8bf*)ap, *(const v8bf*)(ap + 16));

        #pragma unroll
        for (int t = 0; t < 8; ++t) {
            const __bf16* bp = &Bs[(t * 16 + lm) * 32 + half * 8];
            v16bf bfrag = cat16(*(const v8bf*)bp, *(const v8bf*)(bp + 16));
            acc[t] = wmma_bf16(afrag, bfrag, acc[t]);
        }
    }

    #pragma unroll
    for (int t = 0; t < 8; ++t) {
        #pragma unroll
        for (int r = 0; r < 8; ++r) {
            int grow = row0 + wave * 16 + half * 8 + r;
            int gcol = col0 + t * 16 + lm;
            out[(size_t)grow * HIDDEN_ + gcol] = acc[t][r];
        }
    }
}

// =====================================================================
extern "C" void kernel_launch(void* const* d_in, const int* in_sizes, int n_in,
                              void* d_out, int out_size, void* d_ws, size_t ws_size,
                              hipStream_t stream) {
    (void)in_sizes; (void)n_in; (void)out_size; (void)ws_size;
    const float* hidden = (const float*)d_in[0];
    const float* w_qkv  = (const float*)d_in[1];
    const float* w_out  = (const float*)d_in[2];
    // d_in[3] = k_cache (zeros), d_in[4] = v_cache (zeros): scatter target is
    // replicated in workspace (inputs must not be mutated).
    const int* bt = (const int*)d_in[5];
    float* out = (float*)d_out;

    char* ws = (char*)d_ws;
    __bf16* q_ws    = (__bf16*)(ws);                          // 32 MB  [T,4096]
    __bf16* k_ws    = (__bf16*)(ws + ((size_t)32 << 20));     //  8 MB  paged K
    __bf16* v_ws    = (__bf16*)(ws + ((size_t)40 << 20));     //  8 MB  paged V
    __bf16* attn_ws = (__bf16*)(ws + ((size_t)48 << 20));     // 32 MB  [T,4096]

    qkv_gemm_kernel<<<dim3(NQKV_ / 128, T_ / 128), 256, 0, stream>>>(
        hidden, w_qkv, bt, q_ws, k_ws, v_ws);

    attn_kernel<<<dim3(S_ / 16, B_ * HKV_), 128, 0, stream>>>(
        q_ws, k_ws, v_ws, bt, attn_ws);

    out_gemm_kernel<<<dim3(HIDDEN_ / 128, T_ / 128), 256, 0, stream>>>(
        attn_ws, w_out, out);
}
